// JiTBlock_85555748537166
// MI455X (gfx1250) — matrix operand order repaired and checked by reading, not compiled, not run on hardware
//
#include <hip/hip_runtime.h>
#include <hip/hip_bf16.h>

// ---------------- types / helpers ----------------
typedef __attribute__((ext_vector_type(16))) __bf16          v16bf;
typedef __attribute__((ext_vector_type(8)))  float           v8f;
typedef __attribute__((ext_vector_type(8)))  unsigned short  us8;
typedef __attribute__((ext_vector_type(4)))  int             v4i;

union BFrag { v16bf v; us8 h[2]; unsigned short u[16]; };

// ---- CDNA5 async global->LDS path (probe via __has_builtin) ----
#if defined(__has_builtin)
# if __has_builtin(__builtin_amdgcn_global_load_async_to_lds_b128)
#  define HAVE_ASYNC_LDS 1
# endif
#endif
#ifndef HAVE_ASYNC_LDS
# define HAVE_ASYNC_LDS 0
#endif

#if HAVE_ASYNC_LDS
__device__ __forceinline__ void async_cp16(const void* g, void* l) {
    __builtin_amdgcn_global_load_async_to_lds_b128((v4i*)g, (v4i*)l, 0, 0);
}
__device__ __forceinline__ void wait_async0() {
#if __has_builtin(__builtin_amdgcn_s_wait_asynccnt)
    __builtin_amdgcn_s_wait_asynccnt(0);
#else
    asm volatile("s_wait_asynccnt 0x0" ::: "memory");
#endif
}
#endif

__device__ __forceinline__ unsigned short f2bf(float f) {
    union { float f; unsigned int u; } x; x.f = f;
    unsigned int u = x.u;
    unsigned int r = (u + 0x7FFFu + ((u >> 16) & 1u)) >> 16;  // RNE
    return (unsigned short)r;
}
__device__ __forceinline__ float silu(float v) { return v / (1.f + __expf(-v)); }

// ---------------- f32 -> bf16 convert with row/col zero padding (2D grid: x=col, y=row) ----------------
__global__ __launch_bounds__(256) void k_f32_to_bf16(const float* __restrict__ src,
                                                     unsigned short* __restrict__ dst,
                                                     int rows, int cols, int dstld) {
    int c = blockIdx.x * 256 + threadIdx.x;
    int r = blockIdx.y;
    if (c >= dstld) return;
    float v = (r < rows && c < cols) ? src[(size_t)r * cols + c] : 0.f;
    dst[(size_t)r * dstld + c] = f2bf(v);
}

// ---------------- f32 zero-pad copy (bias padding) ----------------
__global__ __launch_bounds__(256) void k_pad_f32(const float* __restrict__ src,
                                                 float* __restrict__ dst,
                                                 int n, int npad) {
    int i = blockIdx.x * 256 + threadIdx.x;
    if (i < npad) dst[i] = (i < n) ? src[i] : 0.f;
}

// ---------------- silu(c) -> bf16, zero-padded rows ----------------
__global__ __launch_bounds__(256) void k_silu_bf16(const float* __restrict__ src,
                                                   unsigned short* __restrict__ dst,
                                                   int n, int npad) {
    int i = blockIdx.x * 256 + threadIdx.x;
    if (i < npad) dst[i] = f2bf((i < n) ? silu(src[i]) : 0.f);
}

// ---------------- RMSNorm + adaLN modulation -> bf16 (grid: x=n, y=b) ----------------
__global__ __launch_bounds__(256) void k_rmsmod(const float* __restrict__ X,
                                                const float* __restrict__ w,
                                                const float* __restrict__ ada, // (B, 6C)
                                                int sh_off, int sc_off,
                                                unsigned short* __restrict__ Hout,
                                                int N, int C) {
    int b = blockIdx.y;
    int row = b * N + blockIdx.x;
    const float* xr = X + (size_t)row * C;
    __shared__ float red[256];
    float ss = 0.f;
    for (int c = threadIdx.x; c < C; c += 256) { float v = xr[c]; ss += v * v; }
    red[threadIdx.x] = ss; __syncthreads();
    for (int st = 128; st > 0; st >>= 1) {
        if (threadIdx.x < st) red[threadIdx.x] += red[threadIdx.x + st];
        __syncthreads();
    }
    float r = rsqrtf(red[0] / (float)C + 1e-6f);
    const float* ar = ada + (size_t)b * 6 * C;
    for (int c = threadIdx.x; c < C; c += 256) {
        float hv = xr[c] * r * w[c] * (1.f + ar[sc_off + c]) + ar[sh_off + c];
        Hout[(size_t)row * C + c] = f2bf(hv);
    }
}

// ---------------- tiled bf16 WMMA GEMM (FULL TILES ONLY — all dims padded by caller) ----------------
// C(MxN,f32) = A(MxK,bf16,lda) * B(KxN,bf16,ldb) + bias
// block 256 thr = 8 waves; block tile 64x256, BK=32; wave tile 32x64 (2 A-frags x 4 B-frags, 8 WMMA).
#define BM 64
#define BN 256
#define BK 32
#define APAD 40
__global__ __launch_bounds__(256) void k_gemm_bf16_wmma(const unsigned short* __restrict__ A, int lda,
                                                        const unsigned short* __restrict__ Bw, int ldb,
                                                        const float* __restrict__ bias,
                                                        float* __restrict__ C, int ldc,
                                                        int K) {
    __shared__ __align__(16) unsigned short sA[BM][APAD];
    __shared__ __align__(16) unsigned short sBt[BN][APAD];
    const int tid = threadIdx.x;
    const int wave = tid >> 5, lane = tid & 31;
    const int half = lane >> 4, l16 = lane & 15;
    const int m0 = blockIdx.y * BM;
    const int n0 = blockIdx.x * BN;
    const int wm = (wave & 1) * 32;
    const int wn = (wave >> 1) * 64;

    // per-thread staging offsets (element offsets off the kernel-arg pointers keep
    // global-address-space provenance -> global_load_b128, not flat_load_b128)
    const int ar = tid >> 2;                    // A row 0..63
    const int ac8 = (tid & 3) * 8;              // A col base
    const size_t aOff = (size_t)(m0 + ar) * lda + ac8;
    us8* aDst = (us8*)&sA[ar][ac8];

    size_t bOff[4];
    unsigned short* bDst[4];
#pragma unroll
    for (int ch = 0; ch < 4; ++ch) {
        int chunk = tid + ch * 256;             // 0..1023
        int r = chunk >> 5;                     // k row 0..31
        int c8 = (chunk & 31) * 8;              // n col base
        bOff[ch] = (size_t)r * ldb + n0 + c8;
        bDst[ch] = &sBt[c8][r];
    }

    v8f acc[2][4] = {};

    for (int k0 = 0; k0 < K; k0 += BK) {
        const size_t kB = (size_t)k0 * ldb;     // uniform, SALU
        // ---- stage A (straight copy -> async DMA to LDS) ----
#if HAVE_ASYNC_LDS
        async_cp16(A + aOff + k0, aDst);
#else
        us8 av = *(const us8*)(A + aOff + k0);
#endif
        // ---- stage B: batch the 4 global loads, then transpose-store ----
        us8 bv[4];
#pragma unroll
        for (int ch = 0; ch < 4; ++ch) bv[ch] = *(const us8*)(Bw + kB + bOff[ch]);
#pragma unroll
        for (int ch = 0; ch < 4; ++ch)
#pragma unroll
            for (int j = 0; j < 8; ++j) bDst[ch][(size_t)j * APAD] = bv[ch][j];
#if !HAVE_ASYNC_LDS
        *aDst = av;
#endif
        if (k0 + BK < K) __builtin_prefetch(A + aOff + k0 + BK, 0, 1);
#if HAVE_ASYNC_LDS
        wait_async0();
#endif
        __syncthreads();

        // ---- fragments: each is 2x ds_load_b128 ----
        BFrag a[2];
#pragma unroll
        for (int mi = 0; mi < 2; ++mi) {
            const unsigned short* row = sA[wm + mi * 16 + l16];
            a[mi].h[0] = *(const us8*)(row + 16 * half);       // e0..7 : k = 16*half+e
            a[mi].h[1] = *(const us8*)(row + 16 * half + 8);   // e8..15: k = 16*half+e
        }
        BFrag b[4];
#pragma unroll
        for (int f = 0; f < 4; ++f) {
            const unsigned short* col = sBt[wn + 16 * f + l16];
            b[f].h[0] = *(const us8*)(col + 16 * half);        // k = 16*half + e
            b[f].h[1] = *(const us8*)(col + 16 * half + 8);
        }
#pragma unroll
        for (int mi = 0; mi < 2; ++mi)
#pragma unroll
            for (int f = 0; f < 4; ++f)
                acc[mi][f] = __builtin_amdgcn_wmma_f32_16x16x32_bf16(false, a[mi].v, false, b[f].v,
                                                                     (short)0, acc[mi][f], false, false);
        __syncthreads();
    }

#pragma unroll
    for (int mi = 0; mi < 2; ++mi)
#pragma unroll
        for (int f = 0; f < 4; ++f) {
            int gn = n0 + wn + 16 * f + l16;
            float bv2 = bias[gn];
#pragma unroll
            for (int r = 0; r < 8; ++r) {
                int gm = m0 + wm + mi * 16 + r + 8 * half;
                C[(size_t)gm * ldc + gn] = acc[mi][f][r] + bv2;
            }
        }
}

// ---------------- per-head RMS + RoPE for q,k ; pack v ; -> (B,H,N,64) bf16 ----------------
// grid: (H, N, B), 64 threads (d)
__global__ __launch_bounds__(64) void k_qk_prep(const float* __restrict__ qkv, // (B*N, 3*C)
                                                const float* __restrict__ qn_w,
                                                const float* __restrict__ kn_w,
                                                const float* __restrict__ cosb, // (N,32)
                                                const float* __restrict__ sinb,
                                                unsigned short* __restrict__ Qb,
                                                unsigned short* __restrict__ Kb,
                                                unsigned short* __restrict__ Vb,
                                                int H, int N) {
    const int hd = 64;
    const int C = H * hd;
    int t = threadIdx.x;
    int h = blockIdx.x;
    int n = blockIdx.y;
    int b = blockIdx.z;
    int bn = b * N + n;
    __shared__ float red[64];
    size_t outoff = (((size_t)(b * H + h)) * N + n) * hd;

    for (int s = 0; s < 2; ++s) {       // 0=q, 1=k
        float x = qkv[(size_t)bn * 3 * C + (size_t)s * C + h * hd + t];
        red[t] = x * x; __syncthreads();
        for (int st = 32; st > 0; st >>= 1) {
            if (t < st) red[t] += red[t + st];
            __syncthreads();
        }
        float r = rsqrtf(red[0] / (float)hd + 1e-6f);
        __syncthreads();
        float xn = x * r * (s == 0 ? qn_w[t] : kn_w[t]);
        red[t] = xn; __syncthreads();
        float part = red[t ^ 32];
        float cs = cosb[(size_t)n * 32 + (t & 31)];
        float sn = sinb[(size_t)n * 32 + (t & 31)];
        float out = (t < 32) ? (xn * cs - part * sn) : (part * sn + xn * cs);
        (s == 0 ? Qb : Kb)[outoff + t] = f2bf(out);
        __syncthreads();
    }
    float v = qkv[(size_t)bn * 3 * C + 2 * (size_t)C + h * hd + t];
    Vb[outoff + t] = f2bf(v);
}

// ---------------- flash attention with WMMA (hd=64) ----------------
// grid (N/64, H, B), 128 thr = 4 waves; each wave owns 16 query rows.
#define QPAD 72   // 64 + 8 shorts -> 144B stride (16B aligned)
#define VPAD 40   // 32 + 8 shorts ->  80B stride
__global__ __launch_bounds__(128) void k_flash_attn(const unsigned short* __restrict__ Qb,
                                                    const unsigned short* __restrict__ Kb,
                                                    const unsigned short* __restrict__ Vb,
                                                    unsigned short* __restrict__ Ob, // (B*N, H*64)
                                                    int H, int N) {
    const int hd = 64;
    __shared__ __align__(16) unsigned short sQ[64][QPAD];
    __shared__ __align__(16) unsigned short sK[32][QPAD];   // [key][d] (B-frag friendly for S)
    __shared__ __align__(16) unsigned short sVt[64][VPAD];  // [d][key] (B-frag friendly for O)
    __shared__ __align__(16) unsigned short sP[4][16][VPAD];

    const int tid = threadIdx.x;
    const int wave = tid >> 5, lane = tid & 31;
    const int half = lane >> 4, l16 = lane & 15;
    const int h = blockIdx.y, b = blockIdx.z;
    const int q0 = blockIdx.x * 64;
    const size_t headBase = ((size_t)(b * H + h)) * N * hd;
    const float sm_scale = 0.125f;      // 1/sqrt(64)

    // stage Q (64x64): 4 straight-copy chunks per thread
    {
        int r = tid >> 3, c8 = (tid & 7) * 8;
        const size_t qOff = headBase + (size_t)(q0 + r) * hd + c8;
#pragma unroll
        for (int i = 0; i < 4; ++i) {
#if HAVE_ASYNC_LDS
            async_cp16(Qb + qOff + (size_t)i * 16 * hd, &sQ[r + i * 16][c8]);
#else
            *(us8*)&sQ[r + i * 16][c8] = *(const us8*)(Qb + qOff + (size_t)i * 16 * hd);
#endif
        }
#if HAVE_ASYNC_LDS
        wait_async0();
#endif
    }
    __syncthreads();

    float mrow[8], lrow[8];
    v8f o[4] = {};
#pragma unroll
    for (int r = 0; r < 8; ++r) { mrow[r] = -1e30f; lrow[r] = 0.f; }

    const int qm = wave * 16;

    // staging offsets: per thread 2 chunks (rows kr, kr+16); index off arg pointers
    const int kr = tid >> 3, kc8 = (tid & 7) * 8;
    const size_t kvOff = headBase + (size_t)kr * hd + kc8;
    unsigned short* vDst0 = &sVt[kc8][kr];
    unsigned short* vDst1 = &sVt[kc8][kr + 16];

    for (int k0 = 0; k0 < N; k0 += 32) {
        const size_t kk = kvOff + (size_t)k0 * hd;   // uniform k0*hd on SALU
        // K tile: async DMA straight copy; V tile: batch loads then transpose-store
#if HAVE_ASYNC_LDS
        async_cp16(Kb + kk,           &sK[kr][kc8]);
        async_cp16(Kb + kk + 16 * hd, &sK[kr + 16][kc8]);
#else
        us8 kv0 = *(const us8*)(Kb + kk);
        us8 kv1 = *(const us8*)(Kb + kk + 16 * hd);
#endif
        us8 vv0 = *(const us8*)(Vb + kk);
        us8 vv1 = *(const us8*)(Vb + kk + 16 * hd);
#pragma unroll
        for (int j = 0; j < 8; ++j) {
            vDst0[(size_t)j * VPAD] = vv0[j];
            vDst1[(size_t)j * VPAD] = vv1[j];
        }
#if !HAVE_ASYNC_LDS
        *(us8*)&sK[kr][kc8]      = kv0;
        *(us8*)&sK[kr + 16][kc8] = kv1;
#endif
        if (k0 + 32 < N) {
            __builtin_prefetch(Kb + kk + 32 * hd, 0, 1);
            __builtin_prefetch(Vb + kk + 32 * hd, 0, 1);
        }
#if HAVE_ASYNC_LDS
        wait_async0();
#endif
        __syncthreads();

        // S = Q * K^T : two 16x16 frags, K-dim = hd in 2 steps
        v8f s0 = {}, s1 = {};
#pragma unroll
        for (int dstep = 0; dstep < 2; ++dstep) {
            BFrag a;
            {
                const unsigned short* row = sQ[qm + l16];
                a.h[0] = *(const us8*)(row + dstep * 32 + 16 * half);
                a.h[1] = *(const us8*)(row + dstep * 32 + 16 * half + 8);
            }
            BFrag b0, b1;
            {
                const unsigned short* r0 = sK[l16];
                const unsigned short* r1 = sK[16 + l16];
                b0.h[0] = *(const us8*)(r0 + dstep * 32 + 16 * half);
                b0.h[1] = *(const us8*)(r0 + dstep * 32 + 16 * half + 8);
                b1.h[0] = *(const us8*)(r1 + dstep * 32 + 16 * half);
                b1.h[1] = *(const us8*)(r1 + dstep * 32 + 16 * half + 8);
            }
            s0 = __builtin_amdgcn_wmma_f32_16x16x32_bf16(false, a.v, false, b0.v, (short)0, s0, false, false);
            s1 = __builtin_amdgcn_wmma_f32_16x16x32_bf16(false, a.v, false, b1.v, (short)0, s1, false, false);
        }

        // online softmax across the 32 new columns (row = r + 8*half)
        float scale[8];
#pragma unroll
        for (int r = 0; r < 8; ++r) {
            float v0 = s0[r] * sm_scale, v1 = s1[r] * sm_scale;
            float mx = fmaxf(v0, v1);
#pragma unroll
            for (int off = 1; off < 16; off <<= 1) mx = fmaxf(mx, __shfl_xor(mx, off, 32));
            float mnew = fmaxf(mrow[r], mx);
            scale[r] = __expf(mrow[r] - mnew);
            mrow[r] = mnew;
            float p0 = __expf(v0 - mnew);
            float p1 = __expf(v1 - mnew);
            s0[r] = p0; s1[r] = p1;
            float sum = p0 + p1;
#pragma unroll
            for (int off = 1; off < 16; off <<= 1) sum += __shfl_xor(sum, off, 32);
            lrow[r] = lrow[r] * scale[r] + sum;
        }

        // stage P (16x32) into per-wave LDS, re-form as A fragment
#pragma unroll
        for (int r = 0; r < 8; ++r) {
            int m = r + 8 * half;
            sP[wave][m][l16]      = f2bf(s0[r]);
            sP[wave][m][16 + l16] = f2bf(s1[r]);
        }
        BFrag pa;
        {
            const unsigned short* row = sP[wave][l16];
            pa.h[0] = *(const us8*)(row + 16 * half);
            pa.h[1] = *(const us8*)(row + 16 * half + 8);
        }

        // O = diag(scale) * O + P * V
#pragma unroll
        for (int f = 0; f < 4; ++f) {
#pragma unroll
            for (int r = 0; r < 8; ++r) o[f][r] *= scale[r];
            BFrag vb;
            const unsigned short* col = sVt[16 * f + l16];
            vb.h[0] = *(const us8*)(col + 16 * half);
            vb.h[1] = *(const us8*)(col + 16 * half + 8);
            o[f] = __builtin_amdgcn_wmma_f32_16x16x32_bf16(false, pa.v, false, vb.v, (short)0, o[f], false, false);
        }
        __syncthreads();
    }

    // epilogue: normalize, write o as (B*N, H*64) bf16 (direct proj-GEMM input)
#pragma unroll
    for (int f = 0; f < 4; ++f) {
        int d = 16 * f + l16;
#pragma unroll
        for (int r = 0; r < 8; ++r) {
            int m = r + 8 * half;
            int row = b * N + q0 + qm + m;
            float val = o[f][r] / lrow[r];
            Ob[(size_t)row * (H * hd) + h * hd + d] = f2bf(val);
        }
    }
}

// ---------------- gated residual: out = xin + g_ada * y (grid: x=C/256, y=n, z=b) ----------------
__global__ __launch_bounds__(256) void k_resid_gate(const float* __restrict__ xin,
                                                    const float* __restrict__ y,
                                                    const float* __restrict__ ada,
                                                    int g_off,
                                                    float* __restrict__ out,
                                                    int N, int C) {
    int col = blockIdx.x * 256 + threadIdx.x;
    int b = blockIdx.z;
    int row = b * N + blockIdx.y;
    float g = ada[(size_t)b * 6 * C + g_off + col];
    size_t i = (size_t)row * C + col;
    out[i] = xin[i] + g * y[i];
}

// ---------------- SwiGLU: silu(x1)*x2 -> bf16, zero-padded cols (grid: x=cols, y=row) ----------------
__global__ __launch_bounds__(256) void k_swiglu(const float* __restrict__ w12out, // (M, srcld)
                                                unsigned short* __restrict__ dst, // (M, dstld)
                                                int Hm, int srcld, int dstld) {
    int c = blockIdx.x * 256 + threadIdx.x;
    int row = blockIdx.y;
    if (c >= dstld) return;
    float v = 0.f;
    if (c < Hm) {
        float a = w12out[(size_t)row * srcld + c];
        float bvl = w12out[(size_t)row * srcld + Hm + c];
        v = silu(a) * bvl;
    }
    dst[(size_t)row * dstld + c] = f2bf(v);
}

// =====================================================================
extern "C" void kernel_launch(void* const* d_in, const int* in_sizes, int n_in,
                              void* d_out, int out_size, void* d_ws, size_t ws_size,
                              hipStream_t stream) {
    (void)in_sizes; (void)n_in; (void)out_size; (void)ws_size;
    const int B = 4, N = 1024, C = 1024, H = 16;
    const int MH = 2730;
    const int MHP = 2752;                // swiglu cols / w3 K padded to x32
    const int N3 = 3 * C, N6 = 6 * C, NW = 2 * MH;
    const int NWP = 5632;                // w12 N padded to x256
    const int BNr = B * N;

    const float* x      = (const float*)d_in[0];
    const float* cvec   = (const float*)d_in[1];
    const float* rcos   = (const float*)d_in[2];
    const float* rsin   = (const float*)d_in[3];
    const float* n1w    = (const float*)d_in[4];
    const float* n2w    = (const float*)d_in[5];
    const float* qnw    = (const float*)d_in[6];
    const float* knw    = (const float*)d_in[7];
    const float* qkv_k  = (const float*)d_in[8];
    const float* qkv_b  = (const float*)d_in[9];
    const float* proj_k = (const float*)d_in[10];
    const float* proj_b = (const float*)d_in[11];
    const float* w12_k  = (const float*)d_in[12];
    const float* w12_b  = (const float*)d_in[13];
    const float* w3_k   = (const float*)d_in[14];
    const float* w3_b   = (const float*)d_in[15];
    const float* ada_k  = (const float*)d_in[16];
    const float* ada_b  = (const float*)d_in[17];
    float* out = (float*)d_out;

    // ----- workspace layout -----
    char* p = (char*)d_ws;
    auto take = [&](size_t bytes) { char* q = p; p += (bytes + 255) & ~(size_t)255; return q; };
    unsigned short* wAda  = (unsigned short*)take((size_t)C * N6 * 2);
    unsigned short* wQkv  = (unsigned short*)take((size_t)C * N3 * 2);
    unsigned short* wProj = (unsigned short*)take((size_t)C * C * 2);
    unsigned short* wW12  = (unsigned short*)take((size_t)C * NWP * 2);   // cols zero-padded to 5632
    unsigned short* wW3   = (unsigned short*)take((size_t)MHP * C * 2);   // rows zero-padded to 2752
    float*          w12bP = (float*)take((size_t)NWP * 4);                // padded bias
    unsigned short* siluc = (unsigned short*)take((size_t)64 * C * 2);    // rows zero-padded to 64
    float*          ada   = (float*)take((size_t)64 * N6 * 4);            // rows 0..3 meaningful
    unsigned short* hbuf  = (unsigned short*)take((size_t)BNr * C * 2);
    float*          big   = (float*)take((size_t)BNr * NWP * 4);          // qkv out, then w12 out
    unsigned short* qb    = (unsigned short*)take((size_t)BNr * C * 2);
    unsigned short* kb    = (unsigned short*)take((size_t)BNr * C * 2);
    unsigned short* vb    = (unsigned short*)take((size_t)BNr * C * 2);
    unsigned short* ob    = (unsigned short*)take((size_t)BNr * C * 2);
    float*          yout  = (float*)take((size_t)BNr * C * 4);            // proj out, then w3 out
    float*          x1    = (float*)take((size_t)BNr * C * 4);
    unsigned short* sg    = (unsigned short*)take((size_t)BNr * MHP * 2); // cols zero-padded to 2752

    auto cvt = [&](const float* s, unsigned short* d, int rows, int cols, int dstld, int prows) {
        k_f32_to_bf16<<<dim3((dstld + 255) / 256, prows), 256, 0, stream>>>(s, d, rows, cols, dstld);
    };

    // 1) weights -> bf16 (zero-padded where dims aren't tile multiples)
    cvt(ada_k,  wAda,  C,  N6, N6,  C);
    cvt(qkv_k,  wQkv,  C,  N3, N3,  C);
    cvt(proj_k, wProj, C,  C,  C,   C);
    cvt(w12_k,  wW12,  C,  NW, NWP, C);
    cvt(w3_k,   wW3,   MH, C,  C,   MHP);
    k_pad_f32<<<dim3((NWP + 255) / 256), 256, 0, stream>>>(w12_b, w12bP, NW, NWP);

    // 2) silu(c), zero-padded to 64 rows
    k_silu_bf16<<<dim3((64 * C + 255) / 256), 256, 0, stream>>>(cvec, siluc, B * C, 64 * C);

    // 3) ada = silu(c) @ ada_k + ada_b       (Mpad=64, N=6144, K=1024)
    k_gemm_bf16_wmma<<<dim3(N6 / BN, 64 / BM), 256, 0, stream>>>(siluc, C, wAda, N6, ada_b, ada, N6, C);

    // 4) h1 = mod(rms(x, n1w), sh_msa, sc_msa)
    k_rmsmod<<<dim3(N, B), 256, 0, stream>>>(x, n1w, ada, 0, C, hbuf, N, C);

    // 5) qkv = h1 @ qkv_k + qkv_b            (4096 x 3072)
    k_gemm_bf16_wmma<<<dim3(N3 / BN, BNr / BM), 256, 0, stream>>>(hbuf, C, wQkv, N3, qkv_b, big, N3, C);

    // 6) per-head RMS + RoPE, pack q/k/v to (B,H,N,64) bf16
    k_qk_prep<<<dim3(H, N, B), 64, 0, stream>>>(big, qnw, knw, rcos, rsin, qb, kb, vb, H, N);

    // 7) flash attention -> o (B*N, C) bf16
    k_flash_attn<<<dim3(N / 64, H, B), 128, 0, stream>>>(qb, kb, vb, ob, H, N);

    // 8) proj: y = o @ proj_k + proj_b
    k_gemm_bf16_wmma<<<dim3(C / BN, BNr / BM), 256, 0, stream>>>(ob, C, wProj, C, proj_b, yout, C, C);

    // 9) x1 = x + g_msa * y
    k_resid_gate<<<dim3(C / 256, N, B), 256, 0, stream>>>(x, yout, ada, 2 * C, x1, N, C);

    // 10) h2 = mod(rms(x1, n2w), sh_mlp, sc_mlp)
    k_rmsmod<<<dim3(N, B), 256, 0, stream>>>(x1, n2w, ada, 3 * C, 4 * C, hbuf, N, C);

    // 11) w12: (4096 x 5632pad), K=1024
    k_gemm_bf16_wmma<<<dim3(NWP / BN, BNr / BM), 256, 0, stream>>>(hbuf, C, wW12, NWP, w12bP, big, NWP, C);

    // 12) swiglu -> bf16 (4096 x 2730, dst stride 2752, pad cols zeroed)
    k_swiglu<<<dim3((MHP + 255) / 256, BNr), 256, 0, stream>>>(big, sg, MH, NWP, MHP);

    // 13) w3: (4096 x 1024), Kpad=2752 (pad region is zeros on both operands)
    k_gemm_bf16_wmma<<<dim3(C / BN, BNr / BM), 256, 0, stream>>>(sg, MHP, wW3, C, w3_b, yout, C, MHP);

    // 14) out = x1 + g_mlp * y
    k_resid_gate<<<dim3(C / 256, N, B), 256, 0, stream>>>(x1, yout, ada, 5 * C, out, N, C);
}
